// ParentSetPredictionModel_69458211111125
// MI455X (gfx1250) — matrix operand order repaired and checked by reading, not compiled, not run on hardware
//
#include <hip/hip_runtime.h>
#include <hip/hip_bf16.h>

// ---------------------------------------------------------------------------
// Parent-set prediction model forward pass for MI455X (gfx1250, wave32, WMMA)
// All GEMMs run on v_wmma_f32_16x16x32_f16 (f16 inputs, f32 accumulate).
// Tile staging uses GLOBAL_LOAD_ASYNC_TO_LDS_B128 (ASYNCcnt path).
// ---------------------------------------------------------------------------

#define DIM   128
#define NH    8
#define KSZ   32
#define HK    256            // NH*KSZ
#define NL    16
#define WF    4
#define NSAMP 512
#define NVAR  50
#define NTOK  (NSAMP * NVAR) // 25600
#define PSETS 19650

#define GEMM_RELU  1
#define GEMM_RESID 2

typedef __attribute__((ext_vector_type(16))) _Float16 v16h;
typedef __attribute__((ext_vector_type(8)))  _Float16 v8h;
typedef __attribute__((ext_vector_type(8)))  float    v8f;
typedef __attribute__((ext_vector_type(4)))  int      v4i;

// ---- gfx1250 async copy to LDS (ASYNCcnt path), with safe fallback --------
#if __has_builtin(__builtin_amdgcn_global_load_async_to_lds_b128)
#define HAVE_ASYNC 1
typedef __attribute__((address_space(1))) v4i Gv4i;  // global int4
typedef __attribute__((address_space(3))) v4i Lv4i;  // LDS int4
static __device__ inline void async_copy_b128(void* dst_lds, const void* src_glob) {
  __builtin_amdgcn_global_load_async_to_lds_b128((Gv4i*)src_glob, (Lv4i*)dst_lds,
                                                 0, 0);
}
#if __has_builtin(__builtin_amdgcn_s_wait_asynccnt)
#define ASYNC_WAIT() __builtin_amdgcn_s_wait_asynccnt(0)
#else
#define ASYNC_WAIT() asm volatile("s_wait_asynccnt 0" ::: "memory")
#endif
#endif

// Build a 16-half fragment from two aligned 8-half groups (16B each).
static __device__ inline v16h frag_join(const _Float16* lo, const _Float16* hi) {
  v8h a = *(const v8h*)lo;
  v8h b = *(const v8h*)hi;
  v16h r;
#pragma unroll
  for (int i = 0; i < 8; ++i) { r[i] = a[i]; r[i + 8] = b[i]; }
  return r;
}

static __device__ inline v8f wmma_f16(v16h a, v16h b, v8f c) {
  // D = A(16x32 f16) * B(32x16 f16) + C(16x16 f32)
  return __builtin_amdgcn_wmma_f32_16x16x32_f16(false, a, false, b, (short)0, c,
                                                false, false);
}

// ---------------------------------------------------------------------------
// Generic tiled GEMM: C[M,N] = A[M,K](f16) @ B[K,N](f16) + bias, f32 accum.
// Block = 256 threads = 8 waves (4 M-waves x 2 N-waves); block tile 128x64,
// each wave owns a 32x32 output (2x2 WMMA tiles). K stepped by 32.
// ---------------------------------------------------------------------------
__global__ __launch_bounds__(256) void gemm_f16(
    const _Float16* __restrict__ A, const _Float16* __restrict__ B,
    const float* __restrict__ bias, float* __restrict__ C32,
    _Float16* __restrict__ C16, int M, int N, int K, int flags) {
  __shared__ _Float16 As[128][40];  // [m][k], padded to dodge bank conflicts
  __shared__ _Float16 Bs[64][40];   // transposed: [n][k] -> contiguous K per col

  const int tid  = threadIdx.x;
  const int lane = tid & 31;
  const int wave = tid >> 5;
  const int wm   = wave & 3;  // M wave index (0..3)
  const int wn   = wave >> 2; // N wave index (0..1)
  const int mblk = blockIdx.x * 128;
  const int nblk = blockIdx.y * 64;

  v8f acc00 = {}, acc01 = {}, acc10 = {}, acc11 = {};

  for (int k0 = 0; k0 < K; k0 += 32) {
    // ---- Stage A tile (128x32 halves) as 16B segments. Rows beyond M are
    // clamped: they only ever feed output rows >= M which are never stored.
#ifdef HAVE_ASYNC
#pragma unroll
    for (int id = tid; id < 512; id += 256) {
      int r = id >> 2, seg = id & 3;
      int gr = mblk + r;
      if (gr >= M) gr = M - 1;
      async_copy_b128(&As[r][seg * 8], A + (size_t)gr * K + k0 + seg * 8);
    }
#else
#pragma unroll
    for (int id = tid; id < 512; id += 256) {
      int r = id >> 2, seg = id & 3;
      int gr = mblk + r;
      if (gr >= M) gr = M - 1;
      *(v8h*)&As[r][seg * 8] = *(const v8h*)(A + (size_t)gr * K + k0 + seg * 8);
    }
#endif
    // ---- Stage B tile transposed (Bs[n][k]): vectorized 8-col read per thread.
    {
      int kk = tid >> 3;  // 0..31 (K row)
      int ng = tid & 7;   // 0..7  (8-col group)
      v8h bv = *(const v8h*)(B + (size_t)(k0 + kk) * N + nblk + ng * 8);
#pragma unroll
      for (int j = 0; j < 8; ++j) Bs[ng * 8 + j][kk] = bv[j];
    }
    // Prefetch next K-slice into cache hierarchy (global_prefetch_b8).
    if (k0 + 32 < K) {
      int pr = mblk + (tid & 127);
      if (pr < M) __builtin_prefetch(&A[(size_t)pr * K + k0 + 32], 0, 1);
      __builtin_prefetch(&B[(size_t)(k0 + 32 + (tid & 31)) * N + nblk], 0, 1);
    }
#ifdef HAVE_ASYNC
    ASYNC_WAIT();
#endif
    __syncthreads();

    const int am    = lane & 15;          // row (A) / col (B) within tile
    const int khalf = (lane >> 4) * 16;   // A: lanes 16-31 hold K=16..31
    const int koff  = (lane >> 4) * 16;   // B: lanes 16-31 hold K=16..31

    v16h a0 = frag_join(&As[wm * 32 + am][khalf],      &As[wm * 32 + am][8 + khalf]);
    v16h a1 = frag_join(&As[wm * 32 + 16 + am][khalf], &As[wm * 32 + 16 + am][8 + khalf]);
    v16h b0 = frag_join(&Bs[wn * 32 + am][koff],       &Bs[wn * 32 + am][koff + 8]);
    v16h b1 = frag_join(&Bs[wn * 32 + 16 + am][koff],  &Bs[wn * 32 + 16 + am][koff + 8]);

    acc00 = wmma_f16(a0, b0, acc00);
    acc01 = wmma_f16(a0, b1, acc01);
    acc10 = wmma_f16(a1, b0, acc10);
    acc11 = wmma_f16(a1, b1, acc11);
    __syncthreads();
  }

  // Epilogue. C/D layout: element v -> row v (+8 for lanes 16-31), col = lane&15.
  const int col = lane & 15;
  const int rb  = (lane >> 4) * 8;
  v8f accs[2][2] = {{acc00, acc01}, {acc10, acc11}};
#pragma unroll
  for (int ti = 0; ti < 2; ++ti)
#pragma unroll
    for (int tj = 0; tj < 2; ++tj)
#pragma unroll
      for (int v = 0; v < 8; ++v) {
        int r = mblk + wm * 32 + ti * 16 + rb + v;
        int c = nblk + wn * 32 + tj * 16 + col;
        if (r < M) {
          float val = accs[ti][tj][v];
          if (bias) val += bias[c];
          if (flags & GEMM_RELU) val = fmaxf(val, 0.0f);
          size_t o = (size_t)r * N + c;
          if (C32) C32[o] = (flags & GEMM_RESID) ? (C32[o] + val) : val;
          if (C16) C16[o] = (_Float16)val;
        }
      }
}

// ---------------------------------------------------------------------------
// Attention: one block = 4 waves on one (batch, head); K staged row-major and
// V staged transposed in LDS (clamped addresses -> branch-free fragments).
// Scores: one WMMA per 16-key tile (K=32 = head dim). Softmax in LDS with
// wave32 shfl reductions. Out = P@V: two WMMAs per 32-key chunk.
// ---------------------------------------------------------------------------
__global__ __launch_bounds__(128) void attn_f16(
    const _Float16* __restrict__ Q, const _Float16* __restrict__ Km,
    const _Float16* __restrict__ V, _Float16* __restrict__ O, int T, int Tpad) {
  extern __shared__ _Float16 smem[];
  // Layout: Ks[Tpad][40] | Vs[32][Tpad+8] | probs[4][16][Tpad]
  _Float16* Ks = smem;
  _Float16* Vs = Ks + (size_t)Tpad * 40;
  _Float16* probs_all = Vs + (size_t)32 * (Tpad + 8);

  const int tid  = threadIdx.x;
  const int lane = tid & 31;
  const int wave = tid >> 5;
  const int bb   = blockIdx.x >> 3;  // batch (NH == 8)
  const int h    = blockIdx.x & 7;
  const int qt   = blockIdx.y * 4 + wave;
  const int q0   = qt * 16;
  const size_t tokbase = (size_t)bb * T;
  _Float16* probs = probs_all + (size_t)wave * 16 * Tpad;

  // ---- stage K rows: Ks[key][0..31]; clamp padded keys (masked in softmax)
#ifdef HAVE_ASYNC
  for (int id = tid; id < Tpad * 4; id += 128) {
    int key = id >> 2, seg = id & 3;
    int gk = (key < T) ? key : (T - 1);
    async_copy_b128(&Ks[key * 40 + seg * 8],
                    Km + (tokbase + gk) * HK + h * KSZ + seg * 8);
  }
#else
  for (int id = tid; id < Tpad * 4; id += 128) {
    int key = id >> 2, seg = id & 3;
    int gk = (key < T) ? key : (T - 1);
    *(v8h*)&Ks[key * 40 + seg * 8] =
        *(const v8h*)(Km + (tokbase + gk) * HK + h * KSZ + seg * 8);
  }
#endif
  // ---- stage V transposed: Vs[feat][key]; padded keys have probability 0
  for (int id = tid; id < Tpad * 4; id += 128) {
    int key = id >> 2, seg = id & 3;
    int gk = (key < T) ? key : (T - 1);
    v8h vv = *(const v8h*)(V + (tokbase + gk) * HK + h * KSZ + seg * 8);
#pragma unroll
    for (int j = 0; j < 8; ++j) Vs[(seg * 8 + j) * (Tpad + 8) + key] = vv[j];
  }
#ifdef HAVE_ASYNC
  ASYNC_WAIT();
#endif
  __syncthreads();

  const int am     = lane & 15;
  const int khalf  = (lane >> 4) * 8;
  const int koff16 = (lane >> 4) * 16;
  const bool active = (q0 < T);

  // ---- scores = (Q Kt) * 1/sqrt(KSZ) into LDS ----
  if (active) {
    v16h qa = {};
    int qrow = q0 + am;
    if (qrow < T) {
      const _Float16* qp = Q + (tokbase + qrow) * HK + h * KSZ;
      qa = frag_join(qp + koff16, qp + koff16 + 8);
    }
    const float scale = 0.17677669529663688f;  // 1/sqrt(32)
    for (int kt = 0; kt * 16 < Tpad; ++kt) {
      const _Float16* kr = Ks + (kt * 16 + am) * 40;  // B frag: lane = key col
      v16h kb = frag_join(kr + koff16, kr + koff16 + 8);
      v8f cz = {};
      v8f s = wmma_f16(qa, kb, cz);
      int row = (lane >> 4) * 8;
#pragma unroll
      for (int v = 0; v < 8; ++v)
        probs[(row + v) * Tpad + kt * 16 + am] = (_Float16)(s[v] * scale);
    }
  }
  __syncthreads();

  // ---- softmax per query row (32 lanes cooperate per row) ----
  if (active) {
    for (int r = 0; r < 16; ++r) {
      float mx = -1e30f;
      for (int j = lane; j < Tpad; j += 32) {
        float s = (j < T) ? (float)probs[r * Tpad + j] : -1e30f;
        mx = fmaxf(mx, s);
      }
#pragma unroll
      for (int off = 16; off > 0; off >>= 1) mx = fmaxf(mx, __shfl_xor(mx, off, 32));
      float sum = 0.0f;
      for (int j = lane; j < Tpad; j += 32) {
        float s = (j < T) ? (float)probs[r * Tpad + j] : -1e30f;
        sum += __expf(s - mx);
      }
#pragma unroll
      for (int off = 16; off > 0; off >>= 1) sum += __shfl_xor(sum, off, 32);
      float inv = 1.0f / sum;
      for (int j = lane; j < Tpad; j += 32) {
        float s = (j < T) ? (float)probs[r * Tpad + j] : -1e30f;
        probs[r * Tpad + j] = (_Float16)(__expf(s - mx) * inv);
      }
    }
  }
  __syncthreads();

  // ---- out = P @ V (two 16-col WMMA accumulators cover head dim 32) ----
  if (active) {
    v8f o0 = {}, o1 = {};
    for (int kt = 0; kt < Tpad; kt += 32) {
      const _Float16* pr = probs + am * Tpad + kt;  // A frag: lane = query row
      v16h pa = frag_join(pr + koff16, pr + koff16 + 8);
      const _Float16* v0r = Vs + (size_t)am * (Tpad + 8) + kt + koff16;
      const _Float16* v1r = Vs + (size_t)(16 + am) * (Tpad + 8) + kt + koff16;
      v16h vb0 = frag_join(v0r, v0r + 8);  // B frag: lane = feature col
      v16h vb1 = frag_join(v1r, v1r + 8);
      o0 = wmma_f16(pa, vb0, o0);
      o1 = wmma_f16(pa, vb1, o1);
    }
    int rb = (lane >> 4) * 8;
#pragma unroll
    for (int v = 0; v < 8; ++v) {
      int r = q0 + rb + v;
      if (r < T) {
        size_t off = (tokbase + r) * HK + h * KSZ;
        O[off + am]      = (_Float16)o0[v];
        O[off + 16 + am] = (_Float16)o1[v];
      }
    }
  }
}

// ---------------------------------------------------------------------------
// LayerNorm kernels (f32 stats, f16 output). Block = 128 threads = one token.
// ---------------------------------------------------------------------------
__global__ __launch_bounds__(128) void ln3_f16(
    const float* __restrict__ z,
    const float* __restrict__ sq, const float* __restrict__ oq,
    const float* __restrict__ sk, const float* __restrict__ ok,
    const float* __restrict__ sv, const float* __restrict__ ov,
    _Float16* __restrict__ outq, _Float16* __restrict__ outk,
    _Float16* __restrict__ outv) {
  __shared__ float red[4];
  const int d = threadIdx.x, lane = d & 31, wave = d >> 5;
  float x = z[(size_t)blockIdx.x * DIM + d];
  float s = x;
#pragma unroll
  for (int off = 16; off > 0; off >>= 1) s += __shfl_xor(s, off, 32);
  if (lane == 0) red[wave] = s;
  __syncthreads();
  float mean = (red[0] + red[1] + red[2] + red[3]) * (1.0f / DIM);
  __syncthreads();
  float dd = x - mean, q = dd * dd;
#pragma unroll
  for (int off = 16; off > 0; off >>= 1) q += __shfl_xor(q, off, 32);
  if (lane == 0) red[wave] = q;
  __syncthreads();
  float var = (red[0] + red[1] + red[2] + red[3]) * (1.0f / DIM);
  float nn = dd * rsqrtf(var + 1e-5f);
  size_t o = (size_t)blockIdx.x * DIM + d;
  outq[o] = (_Float16)(nn * sq[d] + oq[d]);
  outk[o] = (_Float16)(nn * sk[d] + ok[d]);
  outv[o] = (_Float16)(nn * sv[d] + ov[d]);
}

__global__ __launch_bounds__(128) void ln1_f16(
    const float* __restrict__ z, const float* __restrict__ sc,
    const float* __restrict__ oc, _Float16* __restrict__ out) {
  __shared__ float red[4];
  const int d = threadIdx.x, lane = d & 31, wave = d >> 5;
  float x = z[(size_t)blockIdx.x * DIM + d];
  float s = x;
#pragma unroll
  for (int off = 16; off > 0; off >>= 1) s += __shfl_xor(s, off, 32);
  if (lane == 0) red[wave] = s;
  __syncthreads();
  float mean = (red[0] + red[1] + red[2] + red[3]) * (1.0f / DIM);
  __syncthreads();
  float dd = x - mean, q = dd * dd;
#pragma unroll
  for (int off = 16; off > 0; off >>= 1) q += __shfl_xor(q, off, 32);
  if (lane == 0) red[wave] = q;
  __syncthreads();
  float var = (red[0] + red[1] + red[2] + red[3]) * (1.0f / DIM);
  out[(size_t)blockIdx.x * DIM + d] = (_Float16)(dd * rsqrtf(var + 1e-5f) * sc[d] + oc[d]);
}

// Final LN + max over samples -> variable embeddings. z layout [NSAMP,NVAR,DIM].
__global__ __launch_bounds__(128) void final_emb(
    const float* __restrict__ z, const float* __restrict__ sc,
    const float* __restrict__ oc, float* __restrict__ emb) {
  __shared__ float red[4];
  const int v = blockIdx.x, d = threadIdx.x, lane = d & 31, wave = d >> 5;
  float best = -1e30f;
  for (int b = 0; b < NSAMP; ++b) {
    float x = z[((size_t)b * NVAR + v) * DIM + d];
    float s = x;
#pragma unroll
    for (int off = 16; off > 0; off >>= 1) s += __shfl_xor(s, off, 32);
    if (lane == 0) red[wave] = s;
    __syncthreads();
    float mean = (red[0] + red[1] + red[2] + red[3]) * (1.0f / DIM);
    __syncthreads();
    float dd = x - mean, q = dd * dd;
#pragma unroll
    for (int off = 16; off > 0; off >>= 1) q += __shfl_xor(q, off, 32);
    if (lane == 0) red[wave] = q;
    __syncthreads();
    float var = (red[0] + red[1] + red[2] + red[3]) * (1.0f / DIM);
    best = fmaxf(best, dd * rsqrtf(var + 1e-5f) * sc[d] + oc[d]);
    __syncthreads();
  }
  emb[v * DIM + d] = best;
}

// ---------------------------------------------------------------------------
// Small elementwise / utility kernels
// ---------------------------------------------------------------------------
__global__ void embed_k(const float* __restrict__ x, const float* __restrict__ w_in,
                        const float* __restrict__ b_in, float* __restrict__ z) {
  int idx = blockIdx.x * blockDim.x + threadIdx.x;
  if (idx >= NTOK * DIM) return;
  int tok = idx >> 7, d = idx & 127;
  float a = b_in[d];
#pragma unroll
  for (int j = 0; j < 3; ++j) a += x[tok * 3 + j] * w_in[j * DIM + d];
  z[idx] = a;
}

__global__ void transpose_k(const float* __restrict__ in, float* __restrict__ out,
                            int Bdim, int T) {
  int idx = blockIdx.x * blockDim.x + threadIdx.x;
  if (idx >= NTOK * DIM) return;
  int d = idx & 127;
  int bt = idx >> 7;
  int t = bt % T, b = bt / T;
  out[((size_t)t * Bdim + b) * DIM + d] = in[idx];
}

__global__ void cvt_f16_k(const float* __restrict__ in, _Float16* __restrict__ out, int n) {
  int i = blockIdx.x * blockDim.x + threadIdx.x;
  if (i < n) out[i] = (_Float16)in[i];
}

// comb = [broadcast(target emb) | indicators @ emb], f16 [PSETS, 2*DIM]
__global__ void comb_k(const float* __restrict__ emb, const int* __restrict__ ind,
                       const int* __restrict__ tgt, _Float16* __restrict__ comb) {
  int idx = blockIdx.x * blockDim.x + threadIdx.x;
  if (idx >= PSETS * 2 * DIM) return;
  int p = idx >> 8;   // 2*DIM == 256
  int c = idx & 255;
  float val;
  if (c < DIM) {
    val = emb[tgt[0] * DIM + c];
  } else {
    int d = c - DIM;
    val = 0.0f;
    for (int v = 0; v < NVAR; ++v)
      if (ind[p * NVAR + v]) val += emb[v * DIM + d];
  }
  comb[idx] = (_Float16)val;
}

__global__ void logits_k(const _Float16* __restrict__ h2, const float* __restrict__ sw3,
                         const float* __restrict__ sb3, float* __restrict__ logits) {
  int p = blockIdx.x * blockDim.x + threadIdx.x;
  if (p >= PSETS) return;
  float acc = sb3[0];
#pragma unroll
  for (int j = 0; j < 64; ++j) acc += (float)h2[(size_t)p * 64 + j] * sw3[j];
  logits[p] = acc;
}

// Iterative deterministic top-k (ties broken by lower index). k read on device.
__global__ __launch_bounds__(256) void topk_k(float* __restrict__ logits,
                                              const int* __restrict__ k_top,
                                              float* __restrict__ out) {
  __shared__ float sv[256];
  __shared__ int   si[256];
  int k = k_top[0];
  for (int it = 0; it < k; ++it) {
    float bv = -3e38f;
    int bi = 0x7fffffff;
    for (int j = threadIdx.x; j < PSETS; j += 256) {
      float v = logits[j];
      if (v > bv || (v == bv && j < bi)) { bv = v; bi = j; }
    }
    sv[threadIdx.x] = bv; si[threadIdx.x] = bi;
    __syncthreads();
    for (int s = 128; s > 0; s >>= 1) {
      if (threadIdx.x < s) {
        float ov = sv[threadIdx.x + s]; int oi = si[threadIdx.x + s];
        if (ov > sv[threadIdx.x] || (ov == sv[threadIdx.x] && oi < si[threadIdx.x])) {
          sv[threadIdx.x] = ov; si[threadIdx.x] = oi;
        }
      }
      __syncthreads();
    }
    if (threadIdx.x == 0) { out[it] = sv[0]; logits[si[0]] = -3e38f; }
    __syncthreads();
  }
}

// ---------------------------------------------------------------------------
// Host orchestration
// ---------------------------------------------------------------------------
extern "C" void kernel_launch(void* const* d_in, const int* in_sizes, int n_in,
                              void* d_out, int out_size, void* d_ws, size_t ws_size,
                              hipStream_t stream) {
  (void)in_sizes; (void)n_in; (void)out_size; (void)ws_size;
  const float* x     = (const float*)d_in[0];
  const float* w_in  = (const float*)d_in[1];
  const float* b_in  = (const float*)d_in[2];
  const float* lnq_s = (const float*)d_in[3];
  const float* lnq_o = (const float*)d_in[4];
  const float* lnk_s = (const float*)d_in[5];
  const float* lnk_o = (const float*)d_in[6];
  const float* lnv_s = (const float*)d_in[7];
  const float* lnv_o = (const float*)d_in[8];
  const float* wq    = (const float*)d_in[9];
  const float* bq    = (const float*)d_in[10];
  const float* wk    = (const float*)d_in[11];
  const float* bk    = (const float*)d_in[12];
  const float* wv    = (const float*)d_in[13];
  const float* bv    = (const float*)d_in[14];
  const float* wo    = (const float*)d_in[15];
  const float* bo    = (const float*)d_in[16];
  const float* lnm_s = (const float*)d_in[17];
  const float* lnm_o = (const float*)d_in[18];
  const float* w1    = (const float*)d_in[19];
  const float* b1    = (const float*)d_in[20];
  const float* w2    = (const float*)d_in[21];
  const float* b2    = (const float*)d_in[22];
  const float* lnf_s = (const float*)d_in[23];
  const float* lnf_o = (const float*)d_in[24];
  const float* sw1   = (const float*)d_in[25];
  const float* sb1   = (const float*)d_in[26];
  const float* sw2   = (const float*)d_in[27];
  const float* sb2   = (const float*)d_in[28];
  const float* sw3   = (const float*)d_in[29];
  const float* sb3   = (const float*)d_in[30];
  const int* indicators = (const int*)d_in[31];
  const int* target_idx = (const int*)d_in[32];
  const int* k_top      = (const int*)d_in[33];

  // Workspace bump allocator (256B aligned)
  char* wsp = (char*)d_ws;
  auto alloc = [&](size_t bytes) -> void* {
    void* p = (void*)wsp;
    wsp += (bytes + 255) & ~(size_t)255;
    return p;
  };
  float*    z    = (float*)alloc((size_t)NTOK * DIM * 4);
  float*    zt   = (float*)alloc((size_t)NTOK * DIM * 4);
  _Float16* lq   = (_Float16*)alloc((size_t)NTOK * DIM * 2);
  _Float16* lk   = (_Float16*)alloc((size_t)NTOK * DIM * 2);
  _Float16* lv   = (_Float16*)alloc((size_t)NTOK * DIM * 2);
  _Float16* q16  = (_Float16*)alloc((size_t)NTOK * HK * 2);
  _Float16* k16  = (_Float16*)alloc((size_t)NTOK * HK * 2);
  _Float16* v16  = (_Float16*)alloc((size_t)NTOK * HK * 2);
  _Float16* ao16 = (_Float16*)alloc((size_t)NTOK * HK * 2);
  _Float16* h116 = (_Float16*)alloc((size_t)NTOK * (WF * DIM) * 2);
  _Float16* wq16 = (_Float16*)alloc((size_t)NL * DIM * HK * 2);
  _Float16* wk16 = (_Float16*)alloc((size_t)NL * DIM * HK * 2);
  _Float16* wv16 = (_Float16*)alloc((size_t)NL * DIM * HK * 2);
  _Float16* wo16 = (_Float16*)alloc((size_t)NL * HK * DIM * 2);
  _Float16* w116 = (_Float16*)alloc((size_t)NL * DIM * (WF * DIM) * 2);
  _Float16* w216 = (_Float16*)alloc((size_t)NL * (WF * DIM) * DIM * 2);
  _Float16* sw116 = (_Float16*)alloc((size_t)2 * DIM * DIM * 2);
  _Float16* sw216 = (_Float16*)alloc((size_t)DIM * 64 * 2);
  float*    emb   = (float*)alloc((size_t)NVAR * DIM * 4);
  // Scoring-phase buffers alias attention buffers (disjoint in time).
  _Float16* comb16 = q16;           // PSETS*256 f16 (10.1MB) <= NTOK*HK f16
  _Float16* hs1    = k16;           // PSETS*128 f16
  _Float16* hs2    = v16;           // PSETS*64  f16
  float*    logits = (float*)ao16;  // PSETS f32

  auto conv = [&](const float* src, _Float16* dst, int n) {
    cvt_f16_k<<<(n + 255) / 256, 256, 0, stream>>>(src, dst, n);
  };
  conv(wq, wq16, NL * DIM * HK);
  conv(wk, wk16, NL * DIM * HK);
  conv(wv, wv16, NL * DIM * HK);
  conv(wo, wo16, NL * HK * DIM);
  conv(w1, w116, NL * DIM * WF * DIM);
  conv(w2, w216, NL * WF * DIM * DIM);
  conv(sw1, sw116, 2 * DIM * DIM);
  conv(sw2, sw216, DIM * 64);

  auto gemm = [&](const _Float16* A, const _Float16* Bm, const float* bias,
                  float* C32, _Float16* C16, int M, int N, int K, int flags) {
    dim3 g((M + 127) / 128, N / 64);
    gemm_f16<<<g, 256, 0, stream>>>(A, Bm, bias, C32, C16, M, N, K, flags);
  };

  // z = x @ w_in + b_in
  embed_k<<<(NTOK * DIM + 255) / 256, 256, 0, stream>>>(x, w_in, b_in, z);

  float* zc = z;
  float* zn = zt;
  int Bdim = NSAMP, T = NVAR;  // attention axis alternates via physical transpose
  for (int l = 0; l < NL; ++l) {
    ln3_f16<<<NTOK, 128, 0, stream>>>(zc, lnq_s + l * DIM, lnq_o + l * DIM,
                                      lnk_s + l * DIM, lnk_o + l * DIM,
                                      lnv_s + l * DIM, lnv_o + l * DIM, lq, lk, lv);
    gemm(lq, wq16 + (size_t)l * DIM * HK, bq + l * HK, nullptr, q16, NTOK, HK, DIM, 0);
    gemm(lk, wk16 + (size_t)l * DIM * HK, bk + l * HK, nullptr, k16, NTOK, HK, DIM, 0);
    gemm(lv, wv16 + (size_t)l * DIM * HK, bv + l * HK, nullptr, v16, NTOK, HK, DIM, 0);

    int Tpad = (T + 31) & ~31;
    dim3 ag(Bdim * NH, ((T + 15) / 16 + 3) / 4);
    size_t shb = (size_t)(Tpad * 40 + 32 * (Tpad + 8) + 4 * 16 * Tpad) * 2;
    attn_f16<<<ag, 128, shb, stream>>>(q16, k16, v16, ao16, T, Tpad);

    gemm(ao16, wo16 + (size_t)l * HK * DIM, bo + l * DIM, zc, nullptr, NTOK, DIM, HK,
         GEMM_RESID);

    ln1_f16<<<NTOK, 128, 0, stream>>>(zc, lnm_s + l * DIM, lnm_o + l * DIM, lq);
    gemm(lq, w116 + (size_t)l * DIM * WF * DIM, b1 + l * WF * DIM, nullptr, h116,
         NTOK, WF * DIM, DIM, GEMM_RELU);
    gemm(h116, w216 + (size_t)l * WF * DIM * DIM, b2 + l * DIM, zc, nullptr,
         NTOK, DIM, WF * DIM, GEMM_RESID);

    transpose_k<<<(NTOK * DIM + 255) / 256, 256, 0, stream>>>(zc, zn, Bdim, T);
    { float* tmp = zc; zc = zn; zn = tmp; int t2 = Bdim; Bdim = T; T = t2; }
  }
  // After 16 layers (even number of transposes) zc is [NSAMP, NVAR, DIM].

  final_emb<<<NVAR, 128, 0, stream>>>(zc, lnf_s, lnf_o, emb);
  comb_k<<<(PSETS * 2 * DIM + 255) / 256, 256, 0, stream>>>(emb, indicators,
                                                            target_idx, comb16);
  gemm(comb16, sw116, sb1, nullptr, hs1, PSETS, DIM, 2 * DIM, GEMM_RELU);
  gemm(hs1, sw216, sb2, nullptr, hs2, PSETS, 64, DIM, GEMM_RELU);
  logits_k<<<(PSETS + 255) / 256, 256, 0, stream>>>(hs2, sw3, sb3, logits);
  topk_k<<<1, 256, 0, stream>>>(logits, k_top, (float*)d_out);
}
